// MaskedAutoencoderViT_10428180595227
// MI455X (gfx1250) — matrix operand order repaired and checked by reading, not compiled, not run on hardware
//
#include <hip/hip_runtime.h>
#include <hip/hip_bf16.h>
#include <math.h>

typedef __attribute__((ext_vector_type(16))) _Float16 v16h;
typedef __attribute__((ext_vector_type(8)))  _Float16 v8h;
typedef __attribute__((ext_vector_type(8)))  float    v8f;

#define EPI_NONE  0
#define EPI_BIAS  1
#define EPI_RESID 2
#define EPI_PATCH 3
#define EPI_REL   4

// CDNA5 async global->LDS copy (16B per lane), tracked by ASYNCcnt.
// Inline asm: portable across ROCm7.2 / amdgpu-toolchain builtin arities.
__device__ __forceinline__ void async_b128(unsigned lds_off, const void* gaddr) {
  asm volatile("global_load_async_to_lds_b128 %0, %1, off"
               :: "v"(lds_off), "v"(gaddr) : "memory");
}
__device__ __forceinline__ void wait_async0() {
  asm volatile("s_wait_asynccnt 0x0" ::: "memory");
}

#define BS_STRIDE 40   // 32 K-halves padded to 40 (80B) -> conflict-free LDS banks

// ---------------------------------------------------------------------------
// WMMA GEMM: out[M,N] (f32) = A[M,K] * Bt[N,K]^T (+ epilogue)
// A: f16 row-major (or f32 row-major converted on load, AF32=true)
// Bt: f16, N x K row-major (B pre-transposed)
// Block = 8 waves = 256x64 output. All 8 waves share one 64-col B strip,
// staged per 32-K step into LDS via async loads (double buffered). Each wave
// computes a 32x64 block (2x4 WMMA tiles), A reused 4x from registers.
// blockIdx.y = batch (strides in elements).
// ---------------------------------------------------------------------------
template<int EPI, bool AF32>
__global__ __launch_bounds__(256) void gemm_wmma_k(
    const void* __restrict__ Av, const _Float16* __restrict__ Bt,
    float* __restrict__ out, const float* __restrict__ bias,
    const float* __restrict__ e0, const float* __restrict__ e1,
    int M, int Nn, int K,
    long long strideA, long long strideB, long long strideC)
{
  const int bh = blockIdx.y;
  const int nStrips = Nn >> 6;                    // 64-wide output strips
  const int mg = blockIdx.x / nStrips;            // group of 8 M-tiles (256 rows)
  const int tn = blockIdx.x - mg * nStrips;
  const int tid  = (int)threadIdx.x;
  const int tm   = mg * 8 + (tid >> 5);           // this wave's 32-row block
  const int lane = tid & 31;
  const int mn = lane & 15;                       // A row / B col within 16x16 tile
  const int hh = lane >> 4;                       // lane-half selects K sub-block

  __shared__ _Float16 Bs[2][64 * BS_STRIDE];

  // cooperative B loader: thread -> (col, 8-half chunk) of the 64x32 K-tile
  const int lcol = tid >> 2;
  const int lk   = (tid & 3) * 8;
  const _Float16* Bsrc = Bt + (size_t)bh * strideB + (size_t)(tn * 64 + lcol) * K + lk;
  const unsigned ldsA = (unsigned)(uintptr_t)&Bs[0][lcol * BS_STRIDE + lk];
  const unsigned ldsB = (unsigned)(uintptr_t)&Bs[1][lcol * BS_STRIDE + lk];

  async_b128(ldsA, Bsrc);                         // prefetch k0 = 0 into buffer 0

  v8f acc[2][4] = {};

  const float*    A0f = (const float*)Av    + (size_t)bh * strideA + (size_t)(tm * 32 + mn) * K;
  const _Float16* A0h = (const _Float16*)Av + (size_t)bh * strideA + (size_t)(tm * 32 + mn) * K;

  for (int k0 = 0; k0 < K; k0 += 32) {
    const int cur = (k0 >> 5) & 1;
    wait_async0();                                // my tile for this step arrived
    __syncthreads();                              // everyone's arrived; prior reads done
    if (k0 + 32 < K)                              // prefetch next tile into other buffer
      async_b128(cur ? ldsA : ldsB, Bsrc + k0 + 32);

    v16h a[2], b[4];
    if (AF32) {
      v8f l0 = *(const v8f*)(A0f + k0 + hh * 8);
      v8f h0 = *(const v8f*)(A0f + k0 + 16 + hh * 8);
      v8f l1 = *(const v8f*)(A0f + (size_t)16 * K + k0 + hh * 8);
      v8f h1 = *(const v8f*)(A0f + (size_t)16 * K + k0 + 16 + hh * 8);
      #pragma unroll
      for (int i = 0; i < 8; ++i) {
        a[0][i] = (_Float16)l0[i]; a[0][8 + i] = (_Float16)h0[i];
        a[1][i] = (_Float16)l1[i]; a[1][8 + i] = (_Float16)h1[i];
      }
    } else {
      v8h l0 = *(const v8h*)(A0h + k0 + hh * 8);       // K = h*8 .. h*8+7
      v8h h0 = *(const v8h*)(A0h + k0 + 16 + hh * 8);  // K = 16+h*8 ..
      v8h l1 = *(const v8h*)(A0h + (size_t)16 * K + k0 + hh * 8);
      v8h h1 = *(const v8h*)(A0h + (size_t)16 * K + k0 + 16 + hh * 8);
      #pragma unroll
      for (int i = 0; i < 8; ++i) {
        a[0][i] = l0[i]; a[0][8 + i] = h0[i];
        a[1][i] = l1[i]; a[1][8 + i] = h1[i];
      }
    }
    #pragma unroll
    for (int j = 0; j < 4; ++j) {                 // B fragments from LDS (bank-padded)
      const _Float16* bp = &Bs[cur][(j * 16 + mn) * BS_STRIDE + hh * 8];
      v8h b0 = *(const v8h*)bp;
      v8h b1 = *(const v8h*)(bp + 16);
      #pragma unroll
      for (int i = 0; i < 8; ++i) { b[j][i] = b0[i]; b[j][8 + i] = b1[i]; }
    }
    #pragma unroll
    for (int mi = 0; mi < 2; ++mi)
      #pragma unroll
      for (int j = 0; j < 4; ++j)
        acc[mi][j] = __builtin_amdgcn_wmma_f32_16x16x32_f16(
            false, a[mi], false, b[j], (short)0, acc[mi][j], false, false);
  }

  float* Ob = out + (size_t)bh * strideC;
  #pragma unroll
  for (int mi = 0; mi < 2; ++mi) {
    #pragma unroll
    for (int j = 0; j < 4; ++j) {
      const int col = tn * 64 + j * 16 + mn;
      float bv = 0.f;
      if (EPI == EPI_BIAS || EPI == EPI_RESID || EPI == EPI_PATCH) bv = bias[col];
      #pragma unroll
      for (int r = 0; r < 8; ++r) {
        const int row = tm * 32 + mi * 16 + r + hh * 8;  // C/D: VGPR r -> M=r (+8 hi half)
        float v = acc[mi][j][r] + bv;
        if (EPI == EPI_PATCH)                            // + pos_embed[token%1024][col]
          v += e0[(size_t)(row & 1023) * Nn + col];
        if (EPI == EPI_REL)                              // + rel_h + rel_w
          v += e0[((size_t)bh * M + row) * 32 + (col >> 5)]
             + e1[((size_t)bh * M + row) * 32 + (col & 31)];
        const size_t oi = (size_t)row * Nn + col;
        if (EPI == EPI_RESID) Ob[oi] += v; else Ob[oi] = v;
      }
    }
  }
}

// ------------------------- elementwise / prep kernels ----------------------
__global__ void cast_f16_k(const float* __restrict__ in, _Float16* __restrict__ out, long long n) {
  long long i = (long long)blockIdx.x * 256 + threadIdx.x;
  if (i < n) out[i] = (_Float16)in[i];
}

// in: K x N f32 row-major -> out: N x K f16 row-major
__global__ void transpose_f16_k(const float* __restrict__ in, _Float16* __restrict__ out, int K, int N) {
  long long i = (long long)blockIdx.x * 256 + threadIdx.x;
  if (i >= (long long)K * N) return;
  int n = (int)(i / K), k = (int)(i - (long long)n * K);
  out[i] = (_Float16)in[(size_t)k * N + n];
}

// x (4,3,512,512) -> xim (4096 tokens x 768) f16, k = c*256 + py*16 + px
__global__ void im2col_k(const float* __restrict__ x, _Float16* __restrict__ xim) {
  long long i = (long long)blockIdx.x * 256 + threadIdx.x;
  if (i >= 4096LL * 768) return;
  int k = (int)(i % 768); long long t = i / 768;
  int b = (int)(t >> 10), r = (int)(t & 1023), gy = r >> 5, gx = r & 31;
  int c = k >> 8, py = (k >> 4) & 15, px = k & 15;
  xim[i] = (_Float16)x[(((size_t)b * 3 + c) * 512 + (size_t)(gy * 16 + py)) * 512 + (gx * 16 + px)];
}

// layernorm over 768, one block per token, f16 output
__global__ __launch_bounds__(256) void ln_k(const float* __restrict__ h,
    const float* __restrict__ w, const float* __restrict__ b, _Float16* __restrict__ out) {
  const int t = blockIdx.x, tid = threadIdx.x;
  const float* row = h + (size_t)t * 768;
  __shared__ float red[256];
  float s = 0.f;
  for (int i = tid; i < 768; i += 256) s += row[i];
  red[tid] = s; __syncthreads();
  for (int o = 128; o > 0; o >>= 1) { if (tid < o) red[tid] += red[tid + o]; __syncthreads(); }
  const float mean = red[0] / 768.f; __syncthreads();
  float v = 0.f;
  for (int i = tid; i < 768; i += 256) { float d = row[i] - mean; v += d * d; }
  red[tid] = v; __syncthreads();
  for (int o = 128; o > 0; o >>= 1) { if (tid < o) red[tid] += red[tid + o]; __syncthreads(); }
  const float rstd = rsqrtf(red[0] / 768.f + 1e-5f);
  for (int i = tid; i < 768; i += 256)
    out[(size_t)t * 768 + i] = (_Float16)((row[i] - mean) * rstd * w[i] + b[i]);
}

// qkv (4096 x 2304 f32) -> q16 (scaled by 1/8), k16 (48,1024,64), vt16 (48,64,1024)
__global__ void qkv_split_k(const float* __restrict__ qkv,
    _Float16* __restrict__ q16, _Float16* __restrict__ k16, _Float16* __restrict__ vt16) {
  long long i = (long long)blockIdx.x * 256 + threadIdx.x;
  if (i >= 4096LL * 2304) return;
  int col = (int)(i % 2304); long long t = i / 2304;
  int b = (int)(t >> 10), n = (int)(t & 1023);
  int which = col / 768, rem = col - which * 768, hd = rem >> 6, dd = rem & 63;
  int bh = b * 12 + hd;
  float v = qkv[i];
  if (which == 0)      q16[((size_t)bh * 1024 + n) * 64 + dd] = (_Float16)(v * 0.125f);
  else if (which == 1) k16[((size_t)bh * 1024 + n) * 64 + dd] = (_Float16)v;
  else                 vt16[((size_t)bh * 64 + dd) * 1024 + n] = (_Float16)v;
}

// rel bias: out[bh, row, kk] = 8 * dot(q16[bh,row,:], rel_pos[qc-kk+31,:])  (q16 pre-scaled)
__global__ void rel_k(const _Float16* __restrict__ q16, const float* __restrict__ relpos,
                      float* __restrict__ out, int axis) {
  long long i = (long long)blockIdx.x * 256 + threadIdx.x;
  if (i >= 48LL * 1024 * 32) return;
  int kk = (int)(i & 31); long long rest = i >> 5;
  int row = (int)(rest & 1023); int bh = (int)(rest >> 10);
  int qc = axis == 0 ? (row >> 5) : (row & 31);
  const _Float16* qv = q16 + ((size_t)bh * 1024 + row) * 64;
  const float* rv = relpos + (size_t)(qc - kk + 31) * 64;
  float s = 0.f;
  #pragma unroll 8
  for (int j = 0; j < 64; ++j) s += (float)qv[j] * rv[j];
  out[i] = s * 8.0f;
}

// in-place softmax over rows of 1024
__global__ __launch_bounds__(256) void softmax_k(float* __restrict__ attn) {
  float* row = attn + (size_t)blockIdx.x * 1024;
  const int tid = threadIdx.x;
  __shared__ float red[256];
  float m = -3.4e38f;
  for (int i = tid; i < 1024; i += 256) m = fmaxf(m, row[i]);
  red[tid] = m; __syncthreads();
  for (int o = 128; o > 0; o >>= 1) { if (tid < o) red[tid] = fmaxf(red[tid], red[tid + o]); __syncthreads(); }
  m = red[0]; __syncthreads();
  float s = 0.f;
  for (int i = tid; i < 1024; i += 256) { float e = __expf(row[i] - m); row[i] = e; s += e; }
  red[tid] = s; __syncthreads();
  for (int o = 128; o > 0; o >>= 1) { if (tid < o) red[tid] += red[tid + o]; __syncthreads(); }
  const float inv = 1.f / red[0];
  for (int i = tid; i < 1024; i += 256) row[i] *= inv;
}

// attno (48,1024,64 f32) -> tokens (4096 x 768) f16
__global__ void merge_heads_k(const float* __restrict__ attno, _Float16* __restrict__ x16) {
  long long i = (long long)blockIdx.x * 256 + threadIdx.x;
  if (i >= 4096LL * 768) return;
  int c = (int)(i % 768); long long t = i / 768;
  int b = (int)(t >> 10), n = (int)(t & 1023), hd = c >> 6, dd = c & 63;
  x16[i] = (_Float16)attno[(((size_t)(b * 12 + hd)) * 1024 + n) * 64 + dd];
}

// exact gelu, f32 -> f16
__global__ void gelu_k(const float* __restrict__ in, _Float16* __restrict__ out, long long n) {
  long long i = (long long)blockIdx.x * 256 + threadIdx.x;
  if (i < n) { float x = in[i]; out[i] = (_Float16)(0.5f * x * (1.f + erff(x * 0.70710678118f))); }
}

// ---------------------------------------------------------------------------
extern "C" void kernel_launch(void* const* d_in, const int* in_sizes, int n_in,
                              void* d_out, int out_size, void* d_ws, size_t ws_size,
                              hipStream_t stream) {
  (void)in_sizes; (void)n_in; (void)ws_size;
  const float* x      = (const float*)d_in[0];
  const float* conv_w = (const float*)d_in[1];
  const float* conv_b = (const float*)d_in[2];
  const float* pos    = (const float*)d_in[3];
  const float* ln1_w  = (const float*)d_in[4];
  const float* ln1_b  = (const float*)d_in[5];
  const float* qkv_w  = (const float*)d_in[6];
  const float* qkv_b  = (const float*)d_in[7];
  const float* proj_w = (const float*)d_in[8];
  const float* proj_b = (const float*)d_in[9];
  const float* rph    = (const float*)d_in[10];
  const float* rpw    = (const float*)d_in[11];
  const float* ln2_w  = (const float*)d_in[12];
  const float* ln2_b  = (const float*)d_in[13];
  const float* fc1_w  = (const float*)d_in[14];
  const float* fc1_b  = (const float*)d_in[15];
  const float* fc2_w  = (const float*)d_in[16];
  const float* fc2_b  = (const float*)d_in[17];

  const int TOK = 4096, C = 768, NQ = 2304, MLP = 3072, BH = 48, N = 1024;

  // bump allocator over workspace
  char* ws = (char*)d_ws; size_t off = 0;
  auto alloc = [&](size_t bytes) -> void* {
    void* p = ws + off; off += (bytes + 255) & ~(size_t)255; return p;
  };
  float*     h       = (float*)alloc((size_t)TOK * C * 4);
  _Float16*  x16     = (_Float16*)alloc((size_t)TOK * C * 2);        // im2col / LN out / merged heads
  _Float16*  convw16 = (_Float16*)alloc((size_t)C * C * 2);
  _Float16*  qkvw16  = (_Float16*)alloc(4 * (size_t)NQ * C * 2);
  _Float16*  projw16 = (_Float16*)alloc(4 * (size_t)C * C * 2);
  _Float16*  fc1w16  = (_Float16*)alloc(4 * (size_t)MLP * C * 2);
  _Float16*  fc2w16  = (_Float16*)alloc(4 * (size_t)C * MLP * 2);
  float*     qkvbuf  = (float*)alloc((size_t)TOK * NQ * 4);
  _Float16*  q16     = (_Float16*)alloc((size_t)BH * N * 64 * 2);
  _Float16*  k16     = (_Float16*)alloc((size_t)BH * N * 64 * 2);
  _Float16*  vt16    = (_Float16*)alloc((size_t)BH * 64 * N * 2);
  float*     relh    = (float*)alloc((size_t)BH * N * 32 * 4);
  float*     relw    = (float*)alloc((size_t)BH * N * 32 * 4);
  float*     attn    = (float*)alloc((size_t)BH * N * N * 4);
  float*     attno   = (float*)alloc((size_t)BH * N * 64 * 4);
  float*     fbuf    = (float*)alloc((size_t)TOK * MLP * 4);
  _Float16*  m16     = (_Float16*)alloc((size_t)TOK * MLP * 2);

  auto blks = [](long long n) { return (unsigned)((n + 255) / 256); };
  // block = 256x64 output (8 waves x 32 rows, shared 64-col B strip in LDS)
  auto gblk = [](int M, int N) { return (unsigned)((N / 64) * (M / 256)); };

  // ---- weight prep (f32 -> f16, B pre-transposed to NxK) ----
  cast_f16_k<<<blks((long long)C * C), 256, 0, stream>>>(conv_w, convw16, (long long)C * C);
  for (int d = 0; d < 4; ++d) {
    transpose_f16_k<<<blks((long long)C * NQ), 256, 0, stream>>>(qkv_w + (size_t)d * C * NQ, qkvw16 + (size_t)d * NQ * C, C, NQ);
    transpose_f16_k<<<blks((long long)C * C), 256, 0, stream>>>(proj_w + (size_t)d * C * C, projw16 + (size_t)d * C * C, C, C);
    transpose_f16_k<<<blks((long long)C * MLP), 256, 0, stream>>>(fc1_w + (size_t)d * C * MLP, fc1w16 + (size_t)d * MLP * C, C, MLP);
    transpose_f16_k<<<blks((long long)MLP * C), 256, 0, stream>>>(fc2_w + (size_t)d * MLP * C, fc2w16 + (size_t)d * C * MLP, MLP, C);
  }

  // ---- patch embed: im2col + GEMM (+conv_b +pos_embed) ----
  im2col_k<<<blks((long long)TOK * C), 256, 0, stream>>>(x, x16);
  gemm_wmma_k<EPI_PATCH, false><<<dim3(gblk(TOK, C), 1), 256, 0, stream>>>(
      x16, convw16, h, conv_b, pos, nullptr, TOK, C, C, 0, 0, 0);

  for (int d = 0; d < 4; ++d) {
    // attention
    ln_k<<<TOK, 256, 0, stream>>>(h, ln1_w + d * C, ln1_b + d * C, x16);
    gemm_wmma_k<EPI_BIAS, false><<<dim3(gblk(TOK, NQ), 1), 256, 0, stream>>>(
        x16, qkvw16 + (size_t)d * NQ * C, qkvbuf, qkv_b + d * NQ, nullptr, nullptr, TOK, NQ, C, 0, 0, 0);
    qkv_split_k<<<blks((long long)TOK * NQ), 256, 0, stream>>>(qkvbuf, q16, k16, vt16);
    rel_k<<<blks(48LL * 1024 * 32), 256, 0, stream>>>(q16, rph + (size_t)d * 63 * 64, relh, 0);
    rel_k<<<blks(48LL * 1024 * 32), 256, 0, stream>>>(q16, rpw + (size_t)d * 63 * 64, relw, 1);
    // scores: (q*scale) @ k^T + rel biases, batched over 48 heads
    gemm_wmma_k<EPI_REL, false><<<dim3(gblk(N, N), BH), 256, 0, stream>>>(
        q16, k16, attn, nullptr, relh, relw, N, N, 64,
        (long long)N * 64, (long long)N * 64, (long long)N * N);
    softmax_k<<<BH * N, 256, 0, stream>>>(attn);
    // P @ V (A is f32 probabilities, converted on load)
    gemm_wmma_k<EPI_NONE, true><<<dim3(gblk(N, 64), BH), 256, 0, stream>>>(
        attn, vt16, attno, nullptr, nullptr, nullptr, N, 64, N,
        (long long)N * N, (long long)64 * N, (long long)N * 64);
    merge_heads_k<<<blks((long long)TOK * C), 256, 0, stream>>>(attno, x16);
    gemm_wmma_k<EPI_RESID, false><<<dim3(gblk(TOK, C), 1), 256, 0, stream>>>(
        x16, projw16 + (size_t)d * C * C, h, proj_b + d * C, nullptr, nullptr, TOK, C, C, 0, 0, 0);
    // MLP
    ln_k<<<TOK, 256, 0, stream>>>(h, ln2_w + d * C, ln2_b + d * C, x16);
    gemm_wmma_k<EPI_BIAS, false><<<dim3(gblk(TOK, MLP), 1), 256, 0, stream>>>(
        x16, fc1w16 + (size_t)d * MLP * C, fbuf, fc1_b + d * MLP, nullptr, nullptr, TOK, MLP, C, 0, 0, 0);
    gelu_k<<<blks((long long)TOK * MLP), 256, 0, stream>>>(fbuf, m16, (long long)TOK * MLP);
    gemm_wmma_k<EPI_RESID, false><<<dim3(gblk(TOK, C), 1), 256, 0, stream>>>(
        m16, fc2w16 + (size_t)d * C * MLP, h, fc2_b + d * C, nullptr, nullptr, TOK, C, MLP, 0, 0, 0);
  }

  hipMemcpyAsync(d_out, h, (size_t)out_size * sizeof(float), hipMemcpyDeviceToDevice, stream);
}
